// actor_dec_nw_53781580480513
// MI455X (gfx1250) — matrix operand order repaired and checked by reading, not compiled, not run on hardware
//
#include <hip/hip_runtime.h>
#include <hip/hip_bf16.h>
#include <stdint.h>
#include <stddef.h>

// Problem dims (match reference)
#define NS    64      // num_states
#define OD    8       // padded neighbor width
#define HD    256     // hidden
#define NA    16      // actions
#define NB    16384   // batch
#define MT    128     // batch rows per workgroup
#define HSTR  264     // padded bf16 row stride (256 + 8) for LDS tiles
#define SLOPE 0.01f
#define LNEPS 1e-5f

typedef __bf16        v16bf __attribute__((ext_vector_type(16)));
typedef float         v8f   __attribute__((ext_vector_type(8)));
typedef float         fv4   __attribute__((ext_vector_type(4)));
typedef unsigned int  uv4   __attribute__((ext_vector_type(4)));
typedef unsigned int  u32x4 __attribute__((ext_vector_type(4)));
typedef int           i32x4 __attribute__((ext_vector_type(4)));
typedef int           i32x8 __attribute__((ext_vector_type(8)));

// Dynamic-LDS layout (bytes, all 16B aligned)
#define OFF_X     0u        // float  [MT][64]     32768
#define OFF_W1O   32768u    // float  [8][256]      8192
#define OFF_WP    40960u    // float  [256][16]    16384
#define OFF_BEFF  57344u    // float  [256]         1024
#define OFF_H     58368u    // ushort [MT][HSTR]   67584
#define OFF_W2T   125952u   // ushort [256][HSTR] 135168
#define SMEM_BYTES 261120u  // 255 KB  (< 320 KB per workgroup)

union FragU {          // 32 bytes: one WMMA 16-bit operand per lane
  uv4            q[2];
  unsigned short us[16];
  v16bf          v;
};

__device__ __forceinline__ unsigned short f2bf(float f) {
  unsigned u = __float_as_uint(f);
  u += 0x7FFFu + ((u >> 16) & 1u);   // round-to-nearest-even
  return (unsigned short)(u >> 16);
}
__device__ __forceinline__ float leaky(float x) { return x > 0.f ? x : SLOPE * x; }

// Tensor Data Mover: 2D tile (f32 elements) global -> LDS.  D# fields per
// CDNA5 ISA.  Arity hedged across toolchains.
__device__ __forceinline__ void tdm_load_2d(unsigned lds_off, const void* gptr,
                                            unsigned dim0, unsigned dim1,
                                            unsigned tile0, unsigned tile1,
                                            unsigned long long stride0) {
  unsigned long long ga = (unsigned long long)(uintptr_t)gptr;
  u32x4 d0;
  d0[0] = 1u;                                                   // count=1
  d0[1] = lds_off;                                              // lds_addr
  d0[2] = (unsigned)(ga & 0xFFFFFFFFull);                       // global_addr lo
  d0[3] = (unsigned)((ga >> 32) & 0x01FFFFFFull) | (2u << 30);  // addr hi | type=2
  i32x8 d1;
  d1[0] = (int)(2u << 16);                                      // data_size = 4B
  d1[1] = (int)((dim0 & 0xFFFFu) << 16);                        // tensor_dim0 lo
  d1[2] = (int)((dim0 >> 16) | ((dim1 & 0xFFFFu) << 16));       // dim0 hi | dim1 lo
  d1[3] = (int)((dim1 >> 16) | (tile0 << 16));                  // dim1 hi | tile_dim0
  d1[4] = (int)(tile1 & 0xFFFFu);                               // tile_dim1 (dim2=0)
  d1[5] = (int)(stride0 & 0xFFFFFFFFull);                       // dim0_stride lo
  d1[6] = (int)((stride0 >> 32) & 0xFFFFull);                   // dim0_stride hi
  d1[7] = 0;
  i32x4 z4 = {0, 0, 0, 0};
#if __clang_major__ >= 23
  i32x8 z8 = {0, 0, 0, 0, 0, 0, 0, 0};
  __builtin_amdgcn_tensor_load_to_lds(d0, d1, z4, z4, z8, 0);
#else
  __builtin_amdgcn_tensor_load_to_lds(d0, d1, z4, z4, 0);
#endif
}

__global__ __launch_bounds__(256)
void actor_dec_nw_kernel(const float* __restrict__ x,
                         const int* __restrict__ o_idx,
                         const unsigned char* __restrict__ o_mask,
                         const float* __restrict__ W1, const float* __restrict__ b1,
                         const float* __restrict__ g1, const float* __restrict__ be1,
                         const float* __restrict__ W2, const float* __restrict__ b2,
                         const float* __restrict__ g2, const float* __restrict__ be2,
                         const float* __restrict__ Wp, const float* __restrict__ bp,
                         float* __restrict__ out_logit, float* __restrict__ out_lsm) {
  extern __shared__ char smem[];
  float*          sX    = (float*)(smem + OFF_X);
  float*          sW1o  = (float*)(smem + OFF_W1O);
  float*          sWp   = (float*)(smem + OFF_WP);
  float*          sBeff = (float*)(smem + OFF_BEFF);
  unsigned short* sH    = (unsigned short*)(smem + OFF_H);
  unsigned short* sW2T  = (unsigned short*)(smem + OFF_W2T);

  const int st    = blockIdx.y;        // state
  const int btile = blockIdx.x * MT;   // batch tile base
  const int t     = threadIdx.x;
  const int lane  = t & 31;
  const int wv    = t >> 5;

  //========== Stage: TDM async tiles + W2 transpose/convert ==========
  if (wv == 0) {
    tdm_load_2d(OFF_X,   x + (size_t)btile * NS,               NS,      NB, NS,      MT, NS);
    tdm_load_2d(OFF_W1O, W1 + ((size_t)st * 72 + 64) * HD,     8 * HD,  1,  8 * HD,  1,  8 * HD);
    tdm_load_2d(OFF_WP,  Wp + (size_t)st * HD * NA,            HD * NA, 1,  HD * NA, 1,  HD * NA);
    __builtin_amdgcn_s_wait_tensorcnt(0);
  }
  // Effective bias: one-hot row of W1 folded into b1
  sBeff[t] = W1[((size_t)st * 72 + st) * HD + t] + b1[st * HD + t];
  // W2[s] (f32, [k][n]) -> LDS transposed bf16 [n][k]
  const float* W2s = W2 + (size_t)st * HD * HD;
  for (int base = t * 4; base < HD * HD; base += 256 * 4) {
    const int k = base >> 8, n0 = base & 255;
    const fv4 w4 = *(const fv4*)(W2s + base);
    sW2T[(n0 + 0) * HSTR + k] = f2bf(w4.x);
    sW2T[(n0 + 1) * HSTR + k] = f2bf(w4.y);
    sW2T[(n0 + 2) * HSTR + k] = f2bf(w4.z);
    sW2T[(n0 + 3) * HSTR + k] = f2bf(w4.w);
  }
  __builtin_prefetch(x + (size_t)(btile + MT) * NS, 0, 0);  // warm next tile in L2
  __syncthreads();

  //========== Layer 1 (rank-8 VALU) + LN1 -> sH (bf16) ==========
  const int lir = t & 7;          // 8 lanes per batch row
  const int h0  = lir * 32;       // this lane's 32-wide H chunk
  for (int p = 0; p < 4; ++p) {
    const int r = p * 32 + (t >> 3);
    float onb[OD];
#pragma unroll
    for (int j = 0; j < OD; ++j) {
      const int col = o_idx[st * OD + j];
      onb[j] = o_mask[st * OD + j] ? sX[r * NS + col] : 0.f;
    }
    float v[32], sum = 0.f, ss = 0.f;
#pragma unroll
    for (int i = 0; i < 32; ++i) {
      const int h = h0 + i;
      float a = sBeff[h];
#pragma unroll
      for (int j = 0; j < OD; ++j) a = fmaf(onb[j], sW1o[j * HD + h], a);
      a = leaky(a);
      v[i] = a; sum += a; ss += a * a;
    }
    sum += __shfl_xor(sum, 1); sum += __shfl_xor(sum, 2); sum += __shfl_xor(sum, 4);
    ss  += __shfl_xor(ss, 1);  ss  += __shfl_xor(ss, 2);  ss  += __shfl_xor(ss, 4);
    const float mu   = sum * (1.f / HD);
    const float rstd = rsqrtf(ss * (1.f / HD) - mu * mu + LNEPS);
#pragma unroll
    for (int i = 0; i < 32; i += 2) {
      const int h = h0 + i;
      const float a0 = (v[i]     - mu) * rstd * g1[st * HD + h]     + be1[st * HD + h];
      const float a1 = (v[i + 1] - mu) * rstd * g1[st * HD + h + 1] + be1[st * HD + h + 1];
      *(unsigned*)&sH[r * HSTR + h] = (unsigned)f2bf(a0) | ((unsigned)f2bf(a1) << 16);
    }
  }
  __syncthreads();

  //========== Layer 2: 128x256x256 bf16 WMMA GEMM + LN2 ==========
  const int m0   = wv * 16;       // wave's M-subtile
  const int half = lane >> 4;
  const int ln   = lane & 15;
  const v8f vzero = {0.f, 0.f, 0.f, 0.f, 0.f, 0.f, 0.f, 0.f};
  v8f acc[16];
#pragma unroll
  for (int q = 0; q < 16; ++q) acc[q] = vzero;

  for (int k0 = 0; k0 < HD; k0 += 32) {
    FragU af;   // A 16x32 bf16: lane m = ln, K = k0 + half*16 + e (contiguous)
    const unsigned short* arow = sH + (m0 + ln) * HSTR;
    af.q[0] = *(const uv4*)(arow + k0 + half * 16);
    af.q[1] = *(const uv4*)(arow + k0 + half * 16 + 8);
#pragma unroll
    for (int nt = 0; nt < 16; ++nt) {
      FragU bf;  // B 32x16 bf16: lane n = ln, K = k0 + half*16 + e (contiguous)
      const unsigned short* brow = sW2T + (nt * 16 + ln) * HSTR + k0 + half * 16;
      bf.q[0] = *(const uv4*)(brow);
      bf.q[1] = *(const uv4*)(brow + 8);
      acc[nt] = __builtin_amdgcn_wmma_f32_16x16x32_bf16(
          false, af.v, false, bf.v, (short)0, acc[nt], false, false);
    }
  }
  // bias + leaky + LayerNorm directly on the C layout, write bf16 back to sH
  float b2v[16], g2v[16], be2v[16];
#pragma unroll
  for (int nt = 0; nt < 16; ++nt) {
    const int n = nt * 16 + ln;
    b2v[nt] = b2[st * HD + n]; g2v[nt] = g2[st * HD + n]; be2v[nt] = be2[st * HD + n];
  }
#pragma unroll
  for (int r = 0; r < 8; ++r) {
    float sum = 0.f, ss = 0.f;
#pragma unroll
    for (int nt = 0; nt < 16; ++nt) {
      const float vv = leaky(acc[nt][r] + b2v[nt]);
      acc[nt][r] = vv; sum += vv; ss += vv * vv;
    }
    sum += __shfl_xor(sum, 1); sum += __shfl_xor(sum, 2);
    sum += __shfl_xor(sum, 4); sum += __shfl_xor(sum, 8);
    ss  += __shfl_xor(ss, 1);  ss  += __shfl_xor(ss, 2);
    ss  += __shfl_xor(ss, 4);  ss  += __shfl_xor(ss, 8);
    const float mu   = sum * (1.f / HD);
    const float rstd = rsqrtf(ss * (1.f / HD) - mu * mu + LNEPS);
    const int m = m0 + half * 8 + r;
#pragma unroll
    for (int nt = 0; nt < 16; ++nt) {
      const float nv = (acc[nt][r] - mu) * rstd * g2v[nt] + be2v[nt];
      sH[m * HSTR + nt * 16 + ln] = f2bf(nv);
    }
  }

  //========== Head: 128x16x256 WMMA + log_softmax ==========
  v8f acc3 = vzero;
  for (int k0 = 0; k0 < HD; k0 += 32) {
    FragU af;
    const unsigned short* arow = sH + (m0 + ln) * HSTR;
    af.q[0] = *(const uv4*)(arow + k0 + half * 16);
    af.q[1] = *(const uv4*)(arow + k0 + half * 16 + 8);
    FragU bf;
#pragma unroll
    for (int e = 0; e < 16; ++e) {
      const int k = k0 + half * 16 + e;
      bf.us[e] = f2bf(sWp[k * NA + ln]);
    }
    acc3 = __builtin_amdgcn_wmma_f32_16x16x32_bf16(
        false, af.v, false, bf.v, (short)0, acc3, false, false);
  }
  const float bpv = bp[st * NA + ln];
#pragma unroll
  for (int r = 0; r < 8; ++r) {
    const float lg = acc3[r] + bpv;
    float mx = lg;
    mx = fmaxf(mx, __shfl_xor(mx, 1)); mx = fmaxf(mx, __shfl_xor(mx, 2));
    mx = fmaxf(mx, __shfl_xor(mx, 4)); mx = fmaxf(mx, __shfl_xor(mx, 8));
    float se = __expf(lg - mx);
    se += __shfl_xor(se, 1); se += __shfl_xor(se, 2);
    se += __shfl_xor(se, 4); se += __shfl_xor(se, 8);
    const float lsm = (lg - mx) - __logf(se);
    const int bg = btile + m0 + half * 8 + r;
    const size_t oidx = ((size_t)bg * NS + st) * NA + ln;
    out_logit[oidx] = lg;
    out_lsm[oidx]   = lsm;
  }
}

extern "C" void kernel_launch(void* const* d_in, const int* in_sizes, int n_in,
                              void* d_out, int out_size, void* d_ws, size_t ws_size,
                              hipStream_t stream) {
  (void)in_sizes; (void)n_in; (void)out_size; (void)d_ws; (void)ws_size;
  const float*         x      = (const float*)d_in[0];
  const int*           o_idx  = (const int*)d_in[1];
  const unsigned char* o_mask = (const unsigned char*)d_in[2];
  const float* W1  = (const float*)d_in[3];
  const float* b1  = (const float*)d_in[4];
  const float* g1  = (const float*)d_in[5];
  const float* be1 = (const float*)d_in[6];
  const float* W2  = (const float*)d_in[7];
  const float* b2  = (const float*)d_in[8];
  const float* g2  = (const float*)d_in[9];
  const float* be2 = (const float*)d_in[10];
  const float* Wp  = (const float*)d_in[11];
  const float* bp  = (const float*)d_in[12];
  float* out_logit = (float*)d_out;
  float* out_lsm   = out_logit + (size_t)NB * NS * NA;

  dim3 grid(NB / MT, NS);  // 128 x 64 workgroups, one (state, batch-tile) each
  actor_dec_nw_kernel<<<grid, dim3(256), SMEM_BYTES, stream>>>(
      x, o_idx, o_mask, W1, b1, g1, be1, W2, b2, g2, be2, Wp, bp,
      out_logit, out_lsm);
}
